// MultiChannelCNN_78769700209271
// MI455X (gfx1250) — matrix operand order repaired and checked by reading, not compiled, not run on hardware
//
#include <hip/hip_runtime.h>
#include <stdint.h>

// ---------------------------------------------------------------------------
// MultiChannelCNN on MI455X (gfx1250): embed-gather -> im2col GEMM (conv) ->
// relu -> FC GEMM, all matrix math via v_wmma_f32_16x16x32_bf16 (wave32).
// v3: 32x64 wave tiles (1:1 ds_load:wmma), sched_barrier-pinned pipeline
//     (global loads -> ds loads -> wmma), explicit A ping-pong via K-unroll-2,
//     double-buffered LDS with one barrier per 32-K chunk.
// ---------------------------------------------------------------------------

typedef __bf16 bf16x16 __attribute__((ext_vector_type(16)));
typedef float  f32x8   __attribute__((ext_vector_type(8)));

struct B32x8 { uint4 lo, hi; };   // 32 bytes = one 16-element bf16 fragment

static constexpr int VOCAB  = 32000;
static constexpr int EMBED  = 1024;
static constexpr int HIDDEN = 512;
static constexpr int KW     = 3;
static constexpr int SEQ    = 1024;
static constexpr int BATCH  = 4;
static constexpr int ROWS   = SEQ * BATCH;     // 4096 GEMM rows
static constexpr int KCONV  = KW * EMBED;      // 3072
static constexpr int LDP    = 40;              // LDS row stride in halves (80 B)

__device__ __forceinline__ uint16_t f32_to_bf16_rne(float f) {
  union { float f; uint32_t u; } x; x.f = f;
  uint32_t u = x.u;
  u += 0x7FFFu + ((u >> 16) & 1u);             // round-to-nearest-even
  return (uint16_t)(u >> 16);
}

// --------------------------- prep kernels ----------------------------------

__global__ void k_cast_bf16(const float* __restrict__ src,
                            uint16_t* __restrict__ dst, int n) {
  int i = blockIdx.x * blockDim.x + threadIdx.x;
  int stride = gridDim.x * blockDim.x;
  for (; i < n; i += stride) dst[i] = f32_to_bf16_rne(src[i]);
}

// conv_w (H, E, KW) -> Bconv[h, j*EMBED + e]  (N x K row-major, bf16)
__global__ void k_pack_convw(const float* __restrict__ w,
                             uint16_t* __restrict__ dst) {
  const int total = HIDDEN * KCONV;
  int i = blockIdx.x * blockDim.x + threadIdx.x;
  int stride = gridDim.x * blockDim.x;
  for (; i < total; i += stride) {
    int h = i / KCONV, t = i % KCONV;
    int j = t / EMBED, e = t % EMBED;
    dst[i] = f32_to_bf16_rne(w[(h * EMBED + e) * KW + j]);
  }
}

// A1[r, j*EMBED + e] = emb_table[token(s+j-2, b), e]   (token id 0 padding)
__global__ void k_embed_gather(const int* __restrict__ tokens,
                               const float* __restrict__ table,
                               uint16_t* __restrict__ A1) {
  const int r = blockIdx.x;            // 0..4095
  const int s = r >> 2;                // BATCH = 4
  const int b = r & 3;
  for (int j = 0; j < KW; ++j) {
    const int sp = s + j - (KW - 1);
    const int tok = (sp < 0) ? 0 : tokens[sp * BATCH + b];
    const float* srow = table + (size_t)tok * EMBED;
    uint16_t* drow = A1 + (size_t)r * KCONV + j * EMBED;
    for (int e = threadIdx.x; e < EMBED; e += blockDim.x)
      drow[e] = f32_to_bf16_rne(srow[e]);
  }
}

// --------------------------- WMMA GEMM -------------------------------------
// C[M x N] = A[M x K] * B[N x K]^T + bias[n]. 128x128 block tile, 8 waves
// arranged 4(M) x 2(N); each wave owns a 32x64 tile = 2x4 WMMA tiles.
// B chunk (128 x 32 bf16) double-buffered in LDS, padded 80 B row stride.

__device__ __forceinline__ bf16x16 ld_bfrag(const uint16_t (&sBuf)[128 * LDP],
                                            int ntile, int lane, int boff) {
  const uint4* bp = reinterpret_cast<const uint4*>(
      &sBuf[(ntile * 16 + (lane & 15)) * LDP + boff]);
  B32x8 bb; bb.lo = bp[0]; bb.hi = bp[2];
  return __builtin_bit_cast(bf16x16, bb);
}

template<bool RELU_BF16_OUT>
__global__ __launch_bounds__(256)
void k_gemm_wmma(const uint16_t* __restrict__ A, int lda,
                 const uint16_t* __restrict__ Bw, int ldb,
                 const float* __restrict__ bias,
                 float* __restrict__ Cf, uint16_t* __restrict__ Cbf,
                 int ldo, int K) {
  __shared__ __align__(16) uint16_t sB[2][128 * LDP];

  const int lane = threadIdx.x & 31;
  const int wave = threadIdx.x >> 5;
  const int wm   = wave >> 1;                  // 0..3  (M group, 32 rows)
  const int wn   = wave & 1;                   // 0..1  (N group, 64 cols)
  const int n0   = blockIdx.x * 128;
  const int m0w  = blockIdx.y * 128 + wm * 32;

  f32x8 acc[2][4];
  const f32x8 z = {0.f, 0.f, 0.f, 0.f, 0.f, 0.f, 0.f, 0.f};
#pragma unroll
  for (int mi = 0; mi < 2; ++mi)
#pragma unroll
    for (int j = 0; j < 4; ++j) acc[mi][j] = z;

  // --- per-thread staging slots: two 16B segments of the 128x32 B chunk ---
  const int s0 = threadIdx.x, s1 = threadIdx.x + 256;
  const int row0 = s0 >> 2, part0 = s0 & 3;
  const int row1 = s1 >> 2, part1 = s1 & 3;
  const uint16_t* gB0 = Bw + (size_t)(n0 + row0) * ldb + part0 * 8;
  const uint16_t* gB1 = Bw + (size_t)(n0 + row1) * ldb + part1 * 8;
  const int l0 = row0 * LDP + part0 * 8;
  const int l1 = row1 * LDP + part1 * 8;

  // --- per-lane fragment addressing ---
  const uint16_t* aptr0 =
      A + (size_t)(m0w + (lane & 15)) * lda + ((lane >> 4) * 8);
  const uint16_t* aptr1 = aptr0 + (size_t)16 * lda;
  const int boff = (lane >> 4) * 8;            // half-offset in B frag row

  // --- prologue: chunk 0 -> LDS buf 0; A fragments for k=0 -> abA ---
  {
    uint4 p0 = *reinterpret_cast<const uint4*>(gB0);
    uint4 p1 = *reinterpret_cast<const uint4*>(gB1);
    *reinterpret_cast<uint4*>(&sB[0][l0]) = p0;
    *reinterpret_cast<uint4*>(&sB[0][l1]) = p1;
  }
  B32x8 abA[2], abB[2];
  abA[0].lo = *reinterpret_cast<const uint4*>(aptr0);
  abA[0].hi = *reinterpret_cast<const uint4*>(aptr0 + 16);
  abA[1].lo = *reinterpret_cast<const uint4*>(aptr1);
  abA[1].hi = *reinterpret_cast<const uint4*>(aptr1 + 16);

  // one 32-K phase: compute chunk from sB[rb] with curA; stage chunk kn into
  // sB[rb^1]; load A fragments for kn into nxtA.
  auto phase = [&](const B32x8 (&curA)[2], B32x8 (&nxtA)[2], int rb, int kn) {
    // 1) global loads first (latency overlaps everything below)
    uint4 g0 = *reinterpret_cast<const uint4*>(gB0 + kn);
    uint4 g1 = *reinterpret_cast<const uint4*>(gB1 + kn);
    nxtA[0].lo = *reinterpret_cast<const uint4*>(aptr0 + kn);
    nxtA[0].hi = *reinterpret_cast<const uint4*>(aptr0 + kn + 16);
    nxtA[1].lo = *reinterpret_cast<const uint4*>(aptr1 + kn);
    nxtA[1].hi = *reinterpret_cast<const uint4*>(aptr1 + kn + 16);
    __builtin_amdgcn_sched_barrier(0);
    // 2) all B fragments from LDS (pipelined ds_loads, distinct registers)
    bf16x16 bf[4];
#pragma unroll
    for (int j = 0; j < 4; ++j) bf[j] = ld_bfrag(sB[rb], wn * 4 + j, lane, boff);
    __builtin_amdgcn_sched_barrier(0);
    // 3) back-to-back WMMAs
    bf16x16 a0 = __builtin_bit_cast(bf16x16, curA[0]);
    bf16x16 a1 = __builtin_bit_cast(bf16x16, curA[1]);
#pragma unroll
    for (int j = 0; j < 4; ++j) {
      acc[0][j] = __builtin_amdgcn_wmma_f32_16x16x32_bf16(
          false, a0, false, bf[j], (short)0, acc[0][j], false, false);
      acc[1][j] = __builtin_amdgcn_wmma_f32_16x16x32_bf16(
          false, a1, false, bf[j], (short)0, acc[1][j], false, false);
    }
    // 4) stage next chunk (global-load wait lands here, after the WMMAs)
    *reinterpret_cast<uint4*>(&sB[rb ^ 1][l0]) = g0;
    *reinterpret_cast<uint4*>(&sB[rb ^ 1][l1]) = g1;
  };

  // K multiple of 64 (512 and 3072): unroll by two 32-K phases, explicit
  // A-fragment ping-pong (abA <-> abB), one barrier per phase.
  for (int k0 = 0; k0 < K; k0 += 64) {
    __syncthreads();                            // sB[0] ready
    phase(abA, abB, 0, k0 + 32);
    __syncthreads();                            // sB[1] ready
    const int kn2 = (k0 + 64 < K) ? (k0 + 64) : (k0 + 32);  // last: redundant
    phase(abB, abA, 1, kn2);
  }

  // ---- epilogue: bias (+relu/bf16) ----
#pragma unroll
  for (int mi = 0; mi < 2; ++mi) {
#pragma unroll
    for (int j = 0; j < 4; ++j) {
      const int ncol = n0 + (wn * 4 + j) * 16 + (lane & 15);
      const float bs = bias[ncol];
#pragma unroll
      for (int i = 0; i < 8; ++i) {
        const int m = m0w + mi * 16 + (lane >> 4) * 8 + i;
        float v = acc[mi][j][i] + bs;
        if (RELU_BF16_OUT) {
          v = v > 0.f ? v : 0.f;
          Cbf[(size_t)m * ldo + ncol] = f32_to_bf16_rne(v);
        } else {
          Cf[(size_t)m * ldo + ncol] = v;
        }
      }
    }
  }
}

// --------------------------- launcher --------------------------------------

extern "C" void kernel_launch(void* const* d_in, const int* in_sizes, int n_in,
                              void* d_out, int out_size, void* d_ws,
                              size_t ws_size, hipStream_t stream) {
  (void)in_sizes; (void)n_in; (void)out_size; (void)ws_size;
  const int*   tokens = (const int*)d_in[0];
  const float* emb    = (const float*)d_in[1];
  const float* convw  = (const float*)d_in[2];
  const float* convb  = (const float*)d_in[3];
  const float* fcw    = (const float*)d_in[4];
  const float* fcb    = (const float*)d_in[5];
  float*       out    = (float*)d_out;

  uint8_t* ws = (uint8_t*)d_ws;
  size_t off = 0;
  auto carve = [&](size_t bytes) {
    uint8_t* p = ws + off;
    off = (off + bytes + 255) & ~(size_t)255;
    return p;
  };
  uint16_t* fcw_bf = (uint16_t*)carve((size_t)VOCAB * HIDDEN * 2);  // 32.8 MB
  uint16_t* Bconv  = (uint16_t*)carve((size_t)HIDDEN * KCONV * 2);  //  3.1 MB
  uint16_t* A1     = (uint16_t*)carve((size_t)ROWS * KCONV * 2);    // 25.2 MB
  uint16_t* feat   = (uint16_t*)carve((size_t)ROWS * HIDDEN * 2);   //  4.2 MB

  k_cast_bf16<<<4096, 256, 0, stream>>>(fcw, fcw_bf, VOCAB * HIDDEN);
  k_pack_convw<<<2048, 256, 0, stream>>>(convw, Bconv);
  k_embed_gather<<<ROWS, 256, 0, stream>>>(tokens, emb, A1);

  // conv as GEMM: (4096 x 3072) * (512 x 3072)^T, relu, bf16 out
  k_gemm_wmma<true><<<dim3(HIDDEN / 128, ROWS / 128), 256, 0, stream>>>(
      A1, KCONV, Bconv, KCONV, convb, nullptr, feat, HIDDEN, KCONV);

  // FC: (4096 x 512) * (32000 x 512)^T, f32 logits out
  k_gemm_wmma<false><<<dim3(VOCAB / 128, ROWS / 128), 256, 0, stream>>>(
      feat, HIDDEN, fcw_bf, HIDDEN, fcb, out, nullptr, VOCAB, HIDDEN);
}
